// CG_35613868818618
// MI455X (gfx1250) — matrix operand order repaired and checked by reading, not compiled, not run on hardware
//
#include <hip/hip_runtime.h>
#include <hip/hip_bf16.h>

// ---------------------------------------------------------------------------
// GraphMAE / GIN pipeline for MI455X (gfx1250, wave32, WMMA bf16)
// ---------------------------------------------------------------------------

typedef __attribute__((ext_vector_type(16))) __bf16 v16bf;
typedef __attribute__((ext_vector_type(8)))  float  v8f;

#define GN 100000   // nodes
#define GD 128      // hidden
#define GH 256      // MLP hidden
#define GL 2        // layers
#define BN_EPS 1e-5f

__device__ __forceinline__ void atomAddF(float* p, float v) {
    unsafeAtomicAdd(p, v);   // lowers to global_atomic_add_f32 on gfx1250
}

// ------------------------------ masking ------------------------------------
__global__ void mask_kernel(float* __restrict__ x, const int* __restrict__ mnodes,
                            const float* __restrict__ token, int M) {
    int tid = blockIdx.x * blockDim.x + threadIdx.x;
    int m = tid >> 7;
    if (m >= M) return;
    int d = tid & 127;
    x[(long)mnodes[m] * GD + d] = token[d];
}

// --------------------------- edge scatter-add ------------------------------
// 32 lanes per edge, each lane handles 4 consecutive floats (float4 gather,
// 4 hardware fp32 atomics into agg). h matrix (51.2MB) lives in the 192MB L2.
__global__ void scatter_kernel(const float* __restrict__ h, const int* __restrict__ src,
                               const int* __restrict__ dst, float* __restrict__ agg,
                               int nedges) {
    long tid = (long)blockIdx.x * blockDim.x + threadIdx.x;
    int e = (int)(tid >> 5);
    if (e >= nedges) return;
    int part = (int)(tid & 31);
    int s = src[e], d = dst[e];
    const float4 v = *(const float4*)(h + (long)s * GD + part * 4);
    float* p = agg + (long)d * GD + part * 4;
    atomAddF(p + 0, v.x);
    atomAddF(p + 1, v.y);
    atomAddF(p + 2, v.z);
    atomAddF(p + 3, v.w);
}

// ------------------------------ WMMA GEMM ----------------------------------
// Out[N, NC] = transform(A)[N, K] @ W[K, NC]
// MODE 0: a = A0 + A1                      (GIN aggregate:  h + agg)
// MODE 1: a = relu(A0*scl[k] + sft[k])     (fused BatchNorm+ReLU of previous BN)
//
// W staged into LDS pre-packed in WMMA B-fragment order (one aligned 32-byte
// v16bf record per (col-tile, k-tile, lane)) so each v_wmma_f32_16x16x32_bf16
// is fed by a single 32B LDS read.  B fragments are double-buffered in
// registers across the column-tile loop so ds_load -> wmma latency is hidden.
// Stores take an unguarded fast path for full 16-row tiles (wave-uniform
// branch via readfirstlane); only the final partial block pays predication.
// LDS footprint: K*NC*2 = 64KB  (<= 320KB/WGP).
template <int K, int NC, int MODE>
__global__ __launch_bounds__(256) void gemm_kernel(
    const float* __restrict__ A0, const float* __restrict__ A1,
    const float* __restrict__ scl, const float* __restrict__ sft,
    const float* __restrict__ W, float* __restrict__ Out, int nrows) {
    constexpr int KT   = K / 32;    // k-tiles per row strip
    constexpr int NT   = NC / 16;   // 16-column output tiles
    constexpr int RECS = NT * KT * 32;

    __shared__ v16bf Wlds[RECS];    // fragment-packed weights (32B/record)

    // ---- stage + pack weights:  record = (nt*KT + kt)*32 + lane ----
    for (int rec = threadIdx.x; rec < RECS; rec += 256) {
        const int lane = rec & 31;
        const int tile = rec >> 5;
        const int kt   = tile % KT;
        const int nt   = tile / KT;
        const int half = lane >> 4;
        const int n    = nt * 16 + (lane & 15);
        v16bf t;
#pragma unroll
        for (int v = 0; v < 8; ++v) {
            // B 32x16 layout: lane = column; lanes 0-15: K 0..7,16..23; lanes 16-31 shifted by 8
            const int k0 = kt * 32 + ((v < 4) ? 0 : 16) + half * 8 + 2 * (v & 3);
            t[2 * v]     = (__bf16)W[(long)k0 * NC + n];
            t[2 * v + 1] = (__bf16)W[(long)(k0 + 1) * NC + n];
        }
        Wlds[rec] = t;
    }
    __syncthreads();

    const int lane = threadIdx.x & 31;
    const int wave = threadIdx.x >> 5;
    const int r0   = blockIdx.x * 128 + wave * 16;
    const int half = lane >> 4;
    const int l15  = lane & 15;

    // wave-uniform fast-path test: all 16 rows of this strip valid?
    const bool fullTile =
        (__builtin_amdgcn_readfirstlane(r0) + 16) <= nrows;

    // ---- load A fragments (16x32 bf16 each), documented CDNA5 layout ----
    int arow = r0 + l15;
    if (arow >= nrows) arow = nrows - 1;       // clamp (stores are guarded)
    const float* aptr = A0 + (long)arow * K;
    const float* bptr = nullptr;
    if constexpr (MODE == 0) bptr = A1 + (long)arow * K;

    v16bf afrag[KT];
#pragma unroll
    for (int kt = 0; kt < KT; ++kt) {
        v16bf a;
#pragma unroll
        for (int v = 0; v < 8; ++v) {
            // A 16x32 layout: lanes 0-15: VGPR0-3 -> K 0..7, VGPR4-7 -> K 16..23;
            //                 lanes 16-31 shifted by 8.
            const int k = kt * 32 + ((v < 4) ? 0 : 16) + half * 8 + 2 * (v & 3);
            float f0, f1;
            if constexpr (MODE == 0) {
                f0 = aptr[k] + bptr[k];
                f1 = aptr[k + 1] + bptr[k + 1];
            } else {
                f0 = aptr[k] * scl[k] + sft[k];             f0 = f0 > 0.f ? f0 : 0.f;
                f1 = aptr[k + 1] * scl[k + 1] + sft[k + 1]; f1 = f1 > 0.f ? f1 : 0.f;
            }
            a[2 * v]     = (__bf16)f0;
            a[2 * v + 1] = (__bf16)f1;
        }
        afrag[kt] = a;
    }

    // ---- column-tile loop with register double-buffered B fragments ----
    v16bf bfr[2][KT];
#pragma unroll
    for (int kt = 0; kt < KT; ++kt)
        bfr[0][kt] = Wlds[kt * 32 + lane];

#pragma unroll
    for (int nt = 0; nt < NT; ++nt) {
        const int cur = nt & 1;
        const int nxt = cur ^ 1;
        if (nt + 1 < NT) {                       // prefetch next tile's B frags
#pragma unroll
            for (int kt = 0; kt < KT; ++kt)
                bfr[nxt][kt] = Wlds[((nt + 1) * KT + kt) * 32 + lane];
        }

        v8f c = {};
#pragma unroll
        for (int kt = 0; kt < KT; ++kt) {
            c = __builtin_amdgcn_wmma_f32_16x16x32_bf16(
                    false, afrag[kt], false, bfr[cur][kt], (short)0, c, false, false);
        }

        // C/D layout: VGPR r -> row r (lanes 0-15) / row r+8 (lanes 16-31)
        const int n = nt * 16 + l15;
        float* __restrict__ obase = Out + (long)(r0 + half * 8) * NC + n;
        if (fullTile) {
#pragma unroll
            for (int r = 0; r < 8; ++r)
                obase[(long)r * NC] = c[r];
        } else {
#pragma unroll
            for (int r = 0; r < 8; ++r) {
                const int row = r0 + r + half * 8;
                if (row < nrows) Out[(long)row * NC + n] = c[r];
            }
        }
    }
}

// ------------------------ BatchNorm column statistics ----------------------
template <int NC>
__global__ void stats_kernel(const float* __restrict__ Y, float* __restrict__ sum,
                             float* __restrict__ sq, int nrows, int rpb) {
    const int col    = threadIdx.x % NC;
    const int rowoff = threadIdx.x / NC;
    const int rstep  = 256 / NC;
    const long r0    = (long)blockIdx.x * rpb;
    float s = 0.f, q = 0.f;
    for (int r = rowoff; r < rpb; r += rstep) {
        long row = r0 + r;
        if (row >= nrows) break;
        float v = Y[row * NC + col];
        s += v;
        q += v * v;
    }
    atomAddF(&sum[col], s);
    atomAddF(&sq[col], q);
}

template <int NC>
__global__ void bnfin_kernel(const float* __restrict__ sum, const float* __restrict__ sq,
                             const float* __restrict__ g, const float* __restrict__ b,
                             float* __restrict__ scl, float* __restrict__ sft, int nrows) {
    int c = threadIdx.x;
    if (c >= NC) return;
    float inv  = 1.0f / (float)nrows;
    float m    = sum[c] * inv;
    float var  = sq[c] * inv - m * m;           // biased variance (matches ref)
    float s    = g[c] * rsqrtf(var + BN_EPS);
    scl[c] = s;
    sft[c] = b[c] - m * s;
}

// ------------- final BN + ReLU + h store + pool column sums ----------------
__global__ void apply_kernel(const float* __restrict__ Z, const float* __restrict__ scl,
                             const float* __restrict__ sft, float* __restrict__ h,
                             float* __restrict__ pool, int nrows, int rpb) {
    const int col    = threadIdx.x & 127;
    const int rowoff = threadIdx.x >> 7;   // 0/1 : 256 threads, 128 cols
    const long r0    = (long)blockIdx.x * rpb;
    const float s = scl[col], t = sft[col];
    float ps = 0.f;
    for (int r = rowoff; r < rpb; r += 2) {
        long row = r0 + r;
        if (row >= nrows) break;
        float v = Z[row * GD + col] * s + t;
        v = v > 0.f ? v : 0.f;
        h[row * GD + col] = v;
        ps += v;
    }
    atomAddF(&pool[col], ps);
}

// --------------- masked-node cosine loss (one wave per node) ---------------
__global__ void msce_kernel(const float* __restrict__ h1, const float* __restrict__ h2,
                            const int* __restrict__ mnodes, int M, float* __restrict__ acc) {
    const int wave = threadIdx.x >> 5;  // 4 waves / block
    const int lane = threadIdx.x & 31;
    const int m = blockIdx.x * 4 + wave;
    if (m >= M) return;
    const int node = mnodes[m];
    const float4 a = *(const float4*)(h1 + (long)node * GD + lane * 4);
    const float4 b = *(const float4*)(h2 + (long)node * GD + lane * 4);
    float dot = a.x * b.x + a.y * b.y + a.z * b.z + a.w * b.w;
    float na  = a.x * a.x + a.y * a.y + a.z * a.z + a.w * a.w;
    float nb  = b.x * b.x + b.y * b.y + b.z * b.z + b.w * b.w;
    for (int o = 16; o; o >>= 1) {
        dot += __shfl_xor(dot, o);
        na  += __shfl_xor(na, o);
        nb  += __shfl_xor(nb, o);
    }
    if (lane == 0) {
        float denom = fmaxf(sqrtf(na), 1e-12f) * fmaxf(sqrtf(nb), 1e-12f);
        atomAddF(acc, 1.0f - dot / denom);
    }
}

// ----------------------- pooled cosine + final combine ---------------------
__global__ void finloss_kernel(const float* __restrict__ pools, const float* __restrict__ acc,
                               float* __restrict__ out, int M) {
    __shared__ float sdot[8], sna[8], snb[8];
    const int tid = threadIdx.x;                // 256 = L*D
    float x = pools[tid];
    float y = pools[256 + tid];
    float dot = x * y, na = x * x, nb = y * y;
    for (int o = 16; o; o >>= 1) {
        dot += __shfl_xor(dot, o);
        na  += __shfl_xor(na, o);
        nb  += __shfl_xor(nb, o);
    }
    const int lane = tid & 31, wv = tid >> 5;
    if (lane == 0) { sdot[wv] = dot; sna[wv] = na; snb[wv] = nb; }
    __syncthreads();
    if (tid == 0) {
        float D = 0.f, A = 0.f, B = 0.f;
        for (int i = 0; i < 8; ++i) { D += sdot[i]; A += sna[i]; B += snb[i]; }
        float cosv = D / (fmaxf(sqrtf(A), 1e-12f) * fmaxf(sqrtf(B), 1e-12f));
        out[0] = 0.5f * (acc[0] / (float)M) + 0.5f * (1.0f - cosv);
    }
}

// ---------------------------------------------------------------------------
extern "C" void kernel_launch(void* const* d_in, const int* in_sizes, int n_in,
                              void* d_out, int out_size, void* d_ws, size_t ws_size,
                              hipStream_t stream) {
    const float* feat  = (const float*)d_in[0];
    const float* token = (const float*)d_in[1];
    const float* oW1 = (const float*)d_in[2];
    const float* oW2 = (const float*)d_in[3];
    const float* og1 = (const float*)d_in[4];
    const float* ob1 = (const float*)d_in[5];
    const float* og2 = (const float*)d_in[6];
    const float* ob2 = (const float*)d_in[7];
    const float* tW1 = (const float*)d_in[8];
    const float* tW2 = (const float*)d_in[9];
    const float* tg1 = (const float*)d_in[10];
    const float* tb1 = (const float*)d_in[11];
    const float* tg2 = (const float*)d_in[12];
    const float* tb2 = (const float*)d_in[13];
    const int* src = (const int*)d_in[14];
    const int* dst = (const int*)d_in[15];
    const int* mnodes = (const int*)d_in[16];
    const int E = in_sizes[14];
    const int M = in_sizes[16];

    // ---- workspace layout (floats) ----
    const size_t fND = (size_t)GN * GD;   // 12.8M
    const size_t fNH = (size_t)GN * GH;   // 25.6M
    float* ws    = (float*)d_ws;
    float* x     = ws;             // masked features   [N,D]
    float* hO    = x + fND;        // online h          [N,D]
    float* hT    = hO + fND;       // target h          [N,D]
    float* agg   = hT + fND;       // scatter target    [N,D]
    float* Y     = agg + fND;      // MLP hidden        [N,H]
    float* Z     = Y + fNH;        // MLP output        [N,D]
    float* pools = Z + fND;        // [2 * L * D] = 512
    float* sum   = pools + 512;    // [256]
    float* sq    = sum + 256;      // [256]
    float* scl   = sq + 256;       // [256]
    float* sft   = scl + 256;      // [256]
    float* acc   = sft + 256;      // [1]

    hipMemsetAsync(pools, 0, 512 * sizeof(float), stream);
    hipMemsetAsync(acc, 0, sizeof(float), stream);

    // masked input: x = feat; x[mask_nodes] = mask_token
    hipMemcpyAsync(x, feat, fND * sizeof(float), hipMemcpyDeviceToDevice, stream);
    mask_kernel<<<(M * GD + 255) / 256, 256, 0, stream>>>(x, mnodes, token, M);

    const int gemmBlocks    = (GN + 127) / 128;   // 782
    const int RPB           = 512;
    const int statsBlocks   = (GN + RPB - 1) / RPB;
    const int scatterBlocks = (int)(((long)E * 32 + 255) / 256);

    for (int e = 0; e < 2; ++e) {
        const float* hin = (e == 0) ? x : feat;
        const float* W1 = (e == 0) ? oW1 : tW1;
        const float* W2 = (e == 0) ? oW2 : tW2;
        const float* g1 = (e == 0) ? og1 : tg1;
        const float* b1 = (e == 0) ? ob1 : tb1;
        const float* g2 = (e == 0) ? og2 : tg2;
        const float* b2 = (e == 0) ? ob2 : tb2;
        float* h = (e == 0) ? hO : hT;

        for (int l = 0; l < GL; ++l) {
            // GIN aggregate: agg = segment_sum(hin[src], dst)
            hipMemsetAsync(agg, 0, fND * sizeof(float), stream);
            scatter_kernel<<<scatterBlocks, 256, 0, stream>>>(hin, src, dst, agg, E);

            // Y = (hin + agg) @ W1[l]        [N,128]x[128,256]
            gemm_kernel<GD, GH, 0><<<gemmBlocks, 256, 0, stream>>>(
                hin, agg, nullptr, nullptr, W1 + (size_t)l * GD * GH, Y, GN);

            // BatchNorm over Y columns
            hipMemsetAsync(sum, 0, 512 * sizeof(float), stream);  // sum+sq (256 each)
            stats_kernel<GH><<<statsBlocks, 256, 0, stream>>>(Y, sum, sq, GN, RPB);
            bnfin_kernel<GH><<<1, 256, 0, stream>>>(sum, sq, g1 + l * GH, b1 + l * GH,
                                                    scl, sft, GN);

            // Z = relu(BN(Y)) @ W2[l]        [N,256]x[256,128]  (BN+ReLU fused on load)
            gemm_kernel<GH, GD, 1><<<gemmBlocks, 256, 0, stream>>>(
                Y, nullptr, scl, sft, W2 + (size_t)l * GH * GD, Z, GN);

            // BatchNorm over Z columns
            hipMemsetAsync(sum, 0, 512 * sizeof(float), stream);
            stats_kernel<GD><<<statsBlocks, 256, 0, stream>>>(Z, sum, sq, GN, RPB);
            bnfin_kernel<GD><<<1, 128, 0, stream>>>(sum, sq, g2 + l * GD, b2 + l * GD,
                                                    scl, sft, GN);

            // h = relu(BN(Z)); pool[e,l] = column sums of h
            apply_kernel<<<statsBlocks, 256, 0, stream>>>(
                Z, scl, sft, h, pools + e * (GL * GD) + l * GD, GN, RPB);

            hin = h;
        }
    }

    // masked-node SCE (accumulate sum of 1-cos into acc)
    msce_kernel<<<(M + 3) / 4, 128, 0, stream>>>(hO, hT, mnodes, M, acc);

    // pooled SCE + final combine -> d_out[0]
    finloss_kernel<<<1, 256, 0, stream>>>(pools, acc, (float*)d_out, M);
}
